// AQSM_38259568673486
// MI455X (gfx1250) — matrix operand order repaired and checked by reading, not compiled, not run on hardware
//
#include <hip/hip_runtime.h>
#include <math.h>

// ---------------- problem constants ----------------
#define Bz   16
#define Ls   40
#define Cc   256
#define Hh   80
#define Ww   80
#define HWp  6400
#define NQ   10
#define NH   8
#define FFNh 512
#define SEG  8

// ---------------- output layout (floats) ----------------
#define O_O    0                         // [B,NQ,C]   = 40960
#define O_PP   40960                     // [B,NQ,2]   = 320
#define O_GMAP 41280                     // [B,H,W]    = 102400
#define O_ATTN 143680                    // [B,NQ,H,W] = 1024000

// ---------------- scratch layout (bytes, 256-aligned) ----------------
#define WS_POST   ((size_t)0)            // pos_t  [C][HW] f32          6,553,600
#define WS_WKB    ((size_t)6553600)      // wk frag bf16                  131,072
#define WS_WVB    ((size_t)6684672)      // wv frag bf16                  131,072
#define WS_QCA    ((size_t)6815744)      // q_ca [B][16][C] f32           262,144
#define WS_X1     ((size_t)7077888)      // x1 [C] f32                      1,024
#define WS_VHT    ((size_t)7078912)      // vhT [B][C][HW] bf16        52,428,800
#define WS_LOG    ((size_t)59507712)     // logits/probs [B,8,10,HW]   32,768,000
#define WS_CARAW  ((size_t)92275712)     // ca_raw [B][16][C] f32         262,144
#define WS_X4     ((size_t)92537856)     // x4 [B][16][C] f32             262,144
#define WS_PIND   ((size_t)92800000)     // pos_inds [B][NQ] int              640
#define WS_PART   ((size_t)92801024)     // attn@V partials [B][SEG][16][C] f32  2,097,152

typedef __attribute__((ext_vector_type(16))) __bf16 v16bf;
typedef __attribute__((ext_vector_type(8)))  float  v8f;

static __device__ __forceinline__ __bf16 f2bf(float f) { return (__bf16)f; }

static __device__ __forceinline__ v8f wmma_bf16(v16bf a, v16bf b, v8f c) {
  return __builtin_amdgcn_wmma_f32_16x16x32_bf16(false, a, false, b, (short)0, c, false, false);
}

// ================= K1a: sine position embedding, transposed [C][HW] =================
__global__ __launch_bounds__(256) void k_pos(float* posT) {
  int g = blockIdx.x * 256 + threadIdx.x;          // C*HW threads
  int ch = g / HWp, p = g % HWp;
  int i = p / Ww, j = p % Ww;
  const float scale = 6.283185307179586f;
  float yv = (float)(i + 1) / ((float)Hh + 1e-6f) * scale;
  float xv = (float)(j + 1) / ((float)Ww + 1e-6f) * scale;
  float v;
  if (ch < 128) {
    int mm = ch >> 1;
    float tpow = powf(10000.0f, (float)mm / 64.0f);
    float a = yv / tpow;
    v = (ch & 1) ? cosf(a) : sinf(a);
  } else {
    int mm = (ch - 128) >> 1;
    float tpow = powf(10000.0f, (float)mm / 64.0f);
    float a = xv / tpow;
    v = (ch & 1) ? cosf(a) : sinf(a);
  }
  posT[(size_t)ch * HWp + p] = v;
}

// ================= K1b: pack ca_wk / ca_wv into per-lane WMMA B-fragment layout =====
// frag[ (nt*8+ck)*32*16 + lane*16 + e ] = W[nt*16 + lane%16][ ck*32 + (e<8 ? e : e+8) + 8*(lane>=16) ]
__global__ __launch_bounds__(256) void k_wfrag(const float* wk, const float* wvv,
                                               __bf16* wkB, __bf16* wvB) {
  int g = blockIdx.x * 256 + threadIdx.x;          // 2*65536 threads
  int mat = g >> 16; int r = g & 65535;
  int e = r & 15, lane = (r >> 4) & 31, ck = (r >> 9) & 7, nt = (r >> 12) & 15;
  int n = lane & 15, hi = lane >> 4;
  int k = ck * 32 + ((e < 8) ? e : e + 8) + hi * 8;
  int cho = nt * 16 + n;
  float w = (mat ? wvv : wk)[(size_t)cho * Cc + k];
  (mat ? wvB : wkB)[r] = f2bf(w);
}

// ================= K2: constant self-attn output + LN1 -> x1[C] =====================
__global__ __launch_bounds__(256) void k_sax1(const float* sa_wo, const float* sa_bv,
                                              const float* sa_bo, const float* g1,
                                              const float* b1, float* x1) {
  __shared__ float red[256];
  int t = threadIdx.x;
  float s = sa_bo[t];
  for (int i = 0; i < Cc; i++) s += sa_wo[(size_t)t * Cc + i] * sa_bv[i];
  red[t] = s; __syncthreads();
  for (int o = 128; o > 0; o >>= 1) { if (t < o) red[t] += red[t + o]; __syncthreads(); }
  float mu = red[0] / (float)Cc; __syncthreads();
  float d = s - mu;
  red[t] = d * d; __syncthreads();
  for (int o = 128; o > 0; o >>= 1) { if (t < o) red[t] += red[t + o]; __syncthreads(); }
  float var = red[0] / (float)Cc; __syncthreads();
  x1[t] = d * rsqrtf(var + 1e-5f) * g1[t] + b1[t];
}

// ================= K3: qp top-10 + Q projection -> q_ca [B][16][C] ==================
__global__ __launch_bounds__(256) void k_qproj(const float* text, const float* x1,
                                               const float* wq, const float* bq,
                                               float* qca) {
  __shared__ float srow[16][256];
  int b = blockIdx.x, t = threadIdx.x;
  float top[10];
#pragma unroll
  for (int q = 0; q < 10; q++) top[q] = -1e30f;
  for (int lL = 0; lL < Ls; lL++) {
    float v = text[((size_t)b * Ls + lL) * Cc + t];
#pragma unroll
    for (int q = 0; q < 10; q++) {
      if (v > top[q]) { float tmp = top[q]; top[q] = v; v = tmp; }
    }
  }
  float x1v = x1[t];
  for (int q = 0; q < 16; q++) srow[q][t] = x1v + (q < NQ ? top[q] : 0.0f);
  __syncthreads();
  for (int q = 0; q < 16; q++) {
    float s = bq[t];
    for (int i = 0; i < Cc; i++) s += srow[q][i] * wq[(size_t)t * Cc + i];
    qca[((size_t)b * 16 + q) * Cc + t] = s;
  }
}

// ================= K4: fused K/V projection + logits (WMMA) =========================
// grid = B*(HW/64); block = 256 (8 waves). 64 positions = 4 M-tiles per block so each
// weight B-fragment load feeds 4 WMMAs. Wave w: kh/vh N-tiles {w, w+8}; head w logits.
__global__ __launch_bounds__(256) void k_proj_attn(const float* img, const float* posT,
                                                   const __bf16* wkB, const __bf16* wvB,
                                                   const float* bk, const float* bv,
                                                   const float* qca, __bf16* vhT,
                                                   float* logits) {
  __shared__ float  Aimg[32][65];
  __shared__ float  Akp[32][65];
  __shared__ __bf16 khL[64][258];
  int blk = blockIdx.x;
  int b = blk / (HWp / 64), tile = blk % (HWp / 64), p0 = tile * 64;
  int tid = threadIdx.x;
  int wv = tid >> 5, l = tid & 31, m = l & 15, hi = l >> 4;

  // Q fragment for this wave's head (A layout, 16x32 bf16)
  v16bf aq;
#pragma unroll
  for (int e = 0; e < 16; e++) {
    int d = ((e < 8) ? e : e + 8) + hi * 8;
    aq[e] = f2bf(qca[((size_t)b * 16 + m) * Cc + wv * 32 + d]);
  }

  int nt0 = wv, nt1 = wv + 8;
  const v8f zero = {0.f, 0.f, 0.f, 0.f, 0.f, 0.f, 0.f, 0.f};
  v8f aK0[4], aK1[4], aV0[4], aV1[4];
#pragma unroll
  for (int mt = 0; mt < 4; mt++) { aK0[mt] = zero; aK1[mt] = zero; aV0[mt] = zero; aV1[mt] = zero; }

  for (int ck = 0; ck < 8; ck++) {
    { // cooperative tile load: 32 channels x 64 positions (contiguous along p)
      int r0 = tid >> 6, col = tid & 63;
      for (int rr = r0; rr < 32; rr += 4) {
        int ch = ck * 32 + rr;
        float a0 = img[((size_t)b * Cc + ch) * HWp + p0 + col];
        float pz = posT[(size_t)ch * HWp + p0 + col];
        Aimg[rr][col] = a0; Akp[rr][col] = a0 + pz;
      }
    }
    __syncthreads();
    v16bf bk0 = *(const v16bf*)&wkB[(((size_t)nt0 * 8 + ck) * 32 + l) * 16];
    v16bf bk1 = *(const v16bf*)&wkB[(((size_t)nt1 * 8 + ck) * 32 + l) * 16];
    v16bf bv0 = *(const v16bf*)&wvB[(((size_t)nt0 * 8 + ck) * 32 + l) * 16];
    v16bf bv1 = *(const v16bf*)&wvB[(((size_t)nt1 * 8 + ck) * 32 + l) * 16];
#pragma unroll
    for (int mt = 0; mt < 4; mt++) {
      v16bf ai, ap;
#pragma unroll
      for (int e = 0; e < 16; e++) {
        int k = ((e < 8) ? e : e + 8) + hi * 8;
        ai[e] = f2bf(Aimg[k][mt * 16 + m]);
        ap[e] = f2bf(Akp[k][mt * 16 + m]);
      }
      aK0[mt] = wmma_bf16(ap, bk0, aK0[mt]);
      aK1[mt] = wmma_bf16(ap, bk1, aK1[mt]);
      aV0[mt] = wmma_bf16(ai, bv0, aV0[mt]);
      aV1[mt] = wmma_bf16(ai, bv1, aV1[mt]);
    }
    __syncthreads();
  }

  // kh (+bias) -> LDS as bf16 ; vh (+bias) -> global bf16 transposed [C][HW]
  int ch0 = nt0 * 16 + m, ch1 = nt1 * 16 + m;
  float bk0v = bk[ch0], bk1v = bk[ch1], bv0v = bv[ch0], bv1v = bv[ch1];
#pragma unroll
  for (int mt = 0; mt < 4; mt++) {
#pragma unroll
    for (int r = 0; r < 8; r++) {
      int P = mt * 16 + r + hi * 8;
      khL[P][ch0] = f2bf(aK0[mt][r] + bk0v);
      khL[P][ch1] = f2bf(aK1[mt][r] + bk1v);
      vhT[((size_t)b * Cc + ch0) * HWp + p0 + P] = f2bf(aV0[mt][r] + bv0v);
      vhT[((size_t)b * Cc + ch1) * HWp + p0 + P] = f2bf(aV1[mt][r] + bv1v);
    }
  }
  __syncthreads();

  // logits tiles for head wv: A=Q frag, B=kh^T frag from LDS (already bf16)
  const float scale = 0.17677669529663687f;   // 1/sqrt(32)
#pragma unroll
  for (int mt = 0; mt < 4; mt++) {
    v16bf bkf;
#pragma unroll
    for (int e = 0; e < 16; e++) {
      int d = ((e < 8) ? e : e + 8) + hi * 8;
      bkf[e] = khL[mt * 16 + m][wv * 32 + d];
    }
    v8f lg = zero;
    lg = wmma_bf16(aq, bkf, lg);
#pragma unroll
    for (int r = 0; r < 8; r++) {
      int M = r + hi * 8;
      if (M < NQ)
        logits[(((size_t)b * NH + wv) * NQ + M) * HWp + p0 + mt * 16 + m] = lg[r] * scale;
    }
  }
}

// ================= K5: softmax over 6400 positions, in place =========================
__global__ __launch_bounds__(256) void k_softmax(float* logits) {
  __shared__ float red[256];
  float* row = logits + (size_t)blockIdx.x * HWp;
  int t = threadIdx.x;
  float mx = -1e30f;
  for (int p = t; p < HWp; p += 256) mx = fmaxf(mx, row[p]);
  red[t] = mx; __syncthreads();
  for (int o = 128; o > 0; o >>= 1) { if (t < o) red[t] = fmaxf(red[t], red[t + o]); __syncthreads(); }
  mx = red[0]; __syncthreads();
  float s = 0.0f;
  for (int p = t; p < HWp; p += 256) { float e = expf(row[p] - mx); row[p] = e; s += e; }
  red[t] = s; __syncthreads();
  for (int o = 128; o > 0; o >>= 1) { if (t < o) red[t] += red[t + o]; __syncthreads(); }
  float inv = 1.0f / red[0]; __syncthreads();
  for (int p = t; p < HWp; p += 256) row[p] *= inv;
}

// ================= K6: head-average attn (output) + gmap (output) ====================
__global__ __launch_bounds__(256) void k_avg_gmap(const float* probs, float* out) {
  int g = blockIdx.x * 256 + threadIdx.x;          // B*HW threads
  int b = g / HWp, p = g % HWp;
  float mx = -1e30f;
  for (int q = 0; q < NQ; q++) {
    float s = 0.0f;
    for (int h = 0; h < NH; h++) s += probs[(((size_t)b * NH + h) * NQ + q) * HWp + p];
    s *= (1.0f / (float)NH);
    out[O_ATTN + ((size_t)b * NQ + q) * HWp + p] = s;
    mx = fmaxf(mx, s);
  }
  out[O_GMAP + (size_t)b * HWp + p] = mx;
}

// ================= K7a: attn @ V partials (WMMA) over position segments ==============
// grid = B*SEG; wave w = head w, two 16-wide d-tiles, HW/SEG/32 K-chunks per block.
__global__ __launch_bounds__(256) void k_attn_part(const float* probs, const __bf16* vhT,
                                                   float* part) {
  int blk = blockIdx.x;
  int b = blk >> 3, seg = blk & (SEG - 1);
  int tid = threadIdx.x;
  int wv = tid >> 5, l = tid & 31, m = l & 15, hi = l >> 4;
  int mq = (m < NQ) ? m : (NQ - 1);
  const float* arow = probs + (((size_t)b * NH + wv) * NQ + mq) * HWp;
  int ch0 = wv * 32 + m;
  int ch1 = wv * 32 + 16 + m;
  const __bf16* vrow0 = vhT + ((size_t)b * Cc + ch0) * HWp;
  const __bf16* vrow1 = vhT + ((size_t)b * Cc + ch1) * HWp;
  const v8f zero = {0.f, 0.f, 0.f, 0.f, 0.f, 0.f, 0.f, 0.f};
  v8f a0 = zero, a1 = zero;
  int pstart = seg * (HWp / SEG);
  for (int ckp = 0; ckp < (HWp / SEG) / 32; ckp++) {
    int pb = pstart + ckp * 32;
    __builtin_prefetch(vrow0 + pb + 256, 0, 1);
    __builtin_prefetch(vrow1 + pb + 256, 0, 1);
    float4 f0  = *(const float4*)(arow + pb + hi * 16);
    float4 f0b = *(const float4*)(arow + pb + hi * 16 + 4);
    float4 f1  = *(const float4*)(arow + pb + hi * 16 + 8);
    float4 f1b = *(const float4*)(arow + pb + hi * 16 + 12);
    v16bf af;
    af[0]  = f2bf(f0.x);  af[1]  = f2bf(f0.y);  af[2]  = f2bf(f0.z);  af[3]  = f2bf(f0.w);
    af[4]  = f2bf(f0b.x); af[5]  = f2bf(f0b.y); af[6]  = f2bf(f0b.z); af[7]  = f2bf(f0b.w);
    af[8]  = f2bf(f1.x);  af[9]  = f2bf(f1.y);  af[10] = f2bf(f1.z);  af[11] = f2bf(f1.w);
    af[12] = f2bf(f1b.x); af[13] = f2bf(f1b.y); af[14] = f2bf(f1b.z); af[15] = f2bf(f1b.w);
    v16bf b0 = *(const v16bf*)(vrow0 + pb + hi * 16);
    v16bf b1 = *(const v16bf*)(vrow1 + pb + hi * 16);
    a0 = wmma_bf16(af, b0, a0);
    a1 = wmma_bf16(af, b1, a1);
  }
#pragma unroll
  for (int r = 0; r < 8; r++) {
    int M = r + hi * 8;
    part[(((size_t)b * SEG + seg) * 16 + M) * Cc + ch0] = a0[r];
    part[(((size_t)b * SEG + seg) * 16 + M) * Cc + ch1] = a1[r];
  }
}

// ================= K7b: reduce segments + Wo projection -> ca_raw ====================
__global__ __launch_bounds__(256) void k_attn_wo(const float* part, const float* wo,
                                                 const float* bo, float* ca_raw) {
  __shared__ float outH[16][257];
  int b = blockIdx.x, t = threadIdx.x;
  for (int q = 0; q < 16; q++) {
    float s = 0.0f;
    for (int seg = 0; seg < SEG; seg++)
      s += part[(((size_t)b * SEG + seg) * 16 + q) * Cc + t];
    outH[q][t] = s;
  }
  __syncthreads();
  for (int q = 0; q < NQ; q++) {
    float s = bo[t];
    for (int i = 0; i < Cc; i++) s += outH[q][i] * wo[(size_t)t * Cc + i];
    ca_raw[((size_t)b * 16 + q) * Cc + t] = s;
  }
}

// ================= K8: residual + LN2 + FFN + LN3 + post-norm -> x4 ==================
__global__ __launch_bounds__(256) void k_tail(const float* ca_raw, const float* x1,
                                              const float* g2, const float* b2,
                                              const float* g3, const float* b3,
                                              const float* png, const float* pnb,
                                              const float* w1, const float* bb1,
                                              const float* w2, const float* bb2,
                                              float* x4) {
  __shared__ float red[256];
  __shared__ float x2L[256];
  __shared__ float hidL[512];
  int blk = blockIdx.x; int b = blk / NQ, q = blk % NQ; int t = threadIdx.x;
  float x = x1[t] + ca_raw[((size_t)b * 16 + q) * Cc + t];
  // LN2
  red[t] = x; __syncthreads();
  for (int o = 128; o > 0; o >>= 1) { if (t < o) red[t] += red[t + o]; __syncthreads(); }
  float mu = red[0] / (float)Cc; __syncthreads();
  float d = x - mu; red[t] = d * d; __syncthreads();
  for (int o = 128; o > 0; o >>= 1) { if (t < o) red[t] += red[t + o]; __syncthreads(); }
  float var = red[0] / (float)Cc; __syncthreads();
  float x2 = d * rsqrtf(var + 1e-5f) * g2[t] + b2[t];
  x2L[t] = x2; __syncthreads();
  // FFN
  for (int jj = 0; jj < 2; jj++) {
    int j = t + jj * 256;
    float s = bb1[j];
    for (int i = 0; i < Cc; i++) s += x2L[i] * w1[(size_t)j * Cc + i];
    hidL[j] = fmaxf(s, 0.0f);
  }
  __syncthreads();
  float f = bb2[t];
  for (int j = 0; j < FFNh; j++) f += hidL[j] * w2[(size_t)t * FFNh + j];
  float y = x2 + f;
  // LN3
  red[t] = y; __syncthreads();
  for (int o = 128; o > 0; o >>= 1) { if (t < o) red[t] += red[t + o]; __syncthreads(); }
  mu = red[0] / (float)Cc; __syncthreads();
  d = y - mu; red[t] = d * d; __syncthreads();
  for (int o = 128; o > 0; o >>= 1) { if (t < o) red[t] += red[t + o]; __syncthreads(); }
  var = red[0] / (float)Cc; __syncthreads();
  float x3 = d * rsqrtf(var + 1e-5f) * g3[t] + b3[t];
  // post-norm
  red[t] = x3; __syncthreads();
  for (int o = 128; o > 0; o >>= 1) { if (t < o) red[t] += red[t + o]; __syncthreads(); }
  mu = red[0] / (float)Cc; __syncthreads();
  d = x3 - mu; red[t] = d * d; __syncthreads();
  for (int o = 128; o > 0; o >>= 1) { if (t < o) red[t] += red[t + o]; __syncthreads(); }
  var = red[0] / (float)Cc; __syncthreads();
  x4[((size_t)b * 16 + q) * Cc + t] = d * rsqrtf(var + 1e-5f) * png[t] + pnb[t];
}

// ================= K9: greedy distance-guided point selector =========================
__global__ __launch_bounds__(256) void k_select(float* out, int* pinds) {
  __shared__ float mval[HWp];
  __shared__ float rv[256];
  __shared__ int   ri[256];
  __shared__ int   chosen[NQ];
  int b = blockIdx.x, t = threadIdx.x;
  float mind[25];
  for (int i = 0; i < 25; i++) {
    int p = i * 256 + t;
    float g = out[O_GMAP + (size_t)b * HWp + p];
    mval[p] = 1.0f / (1.0f + expf(-g));
  }
  __syncthreads();
  const float norm = 113.13708498984761f;      // sqrt(80^2+80^2)
  const float wdist = 0.1f;
  for (int it = 0; it < NQ; it++) {
    float bvv = -1e30f; int bi = HWp;
    for (int i = 0; i < 25; i++) {
      int p = i * 256 + t;
      float c = mval[p] + ((it == 0) ? 0.0f : wdist * mind[i] / norm);
      if (c > bvv || (c == bvv && p < bi)) { bvv = c; bi = p; }
    }
    rv[t] = bvv; ri[t] = bi; __syncthreads();
    for (int o = 128; o > 0; o >>= 1) {
      if (t < o) {
        float v2 = rv[t + o]; int i2 = ri[t + o];
        if (v2 > rv[t] || (v2 == rv[t] && i2 < ri[t])) { rv[t] = v2; ri[t] = i2; }
      }
      __syncthreads();
    }
    int idx = ri[0];
    if (t == 0) { chosen[it] = idx; mval[idx] = -1e30f; }
    float rr = (float)(idx / Ww), cc = (float)(idx % Ww);
    for (int i = 0; i < 25; i++) {
      int p = i * 256 + t;
      float dr = (float)(p / Ww) - rr, dc = (float)(p % Ww) - cc;
      float dd = sqrtf(dr * dr + dc * dc);
      mind[i] = (it == 0) ? dd : fminf(mind[i], dd);
    }
    __syncthreads();
  }
  if (t < NQ) {
    int ind = chosen[t];
    pinds[b * NQ + t] = ind;
    out[O_PP + ((size_t)b * NQ + t) * 2 + 0] = ((float)(ind % Ww) + 0.5f) / (float)Ww;
    out[O_PP + ((size_t)b * NQ + t) * 2 + 1] = ((float)(ind / Ww) + 0.5f) / (float)Hh;
  }
}

// ================= K10: MLP head on [x4 ; gathered point feature] ====================
__global__ __launch_bounds__(256) void k_head(const float* x4, const int* pinds,
                                              const float* img,
                                              const float* w1, const float* b1,
                                              const float* w2, const float* b2,
                                              const float* w3, const float* b3,
                                              float* out) {
  __shared__ float hc[512];
  __shared__ float o1[256];
  __shared__ float o2[256];
  int blk = blockIdx.x; int b = blk / NQ, q = blk % NQ; int t = threadIdx.x;
  int ind = pinds[b * NQ + q];
  hc[t]       = x4[((size_t)b * 16 + q) * Cc + t];
  hc[256 + t] = img[((size_t)b * Cc + t) * HWp + ind];
  __syncthreads();
  float s = b1[t];
  for (int i = 0; i < 512; i++) s += hc[i] * w1[(size_t)t * 512 + i];
  o1[t] = fmaxf(s, 0.0f); __syncthreads();
  s = b2[t];
  for (int i = 0; i < 256; i++) s += o1[i] * w2[(size_t)t * 256 + i];
  o2[t] = fmaxf(s, 0.0f); __syncthreads();
  s = b3[t];
  for (int i = 0; i < 256; i++) s += o2[i] * w3[(size_t)t * 256 + i];
  out[O_O + ((size_t)b * NQ + q) * Cc + t] = s;
}

// ================================ host launcher ======================================
extern "C" void kernel_launch(void* const* d_in, const int* in_sizes, int n_in,
                              void* d_out, int out_size, void* d_ws, size_t ws_size,
                              hipStream_t stream) {
  (void)in_sizes; (void)n_in; (void)out_size; (void)ws_size;
  const float* text  = (const float*)d_in[0];
  const float* img   = (const float*)d_in[2];
  const float* sa_bv = (const float*)d_in[8];
  const float* sa_wo = (const float*)d_in[9];
  const float* sa_bo = (const float*)d_in[10];
  const float* ca_wq = (const float*)d_in[11];
  const float* ca_bq = (const float*)d_in[12];
  const float* ca_wk = (const float*)d_in[13];
  const float* ca_bk = (const float*)d_in[14];
  const float* ca_wv = (const float*)d_in[15];
  const float* ca_bv = (const float*)d_in[16];
  const float* ca_wo = (const float*)d_in[17];
  const float* ca_bo = (const float*)d_in[18];
  const float* ln1_g = (const float*)d_in[19];
  const float* ln1_b = (const float*)d_in[20];
  const float* ln2_g = (const float*)d_in[21];
  const float* ln2_b = (const float*)d_in[22];
  const float* ln3_g = (const float*)d_in[23];
  const float* ln3_b = (const float*)d_in[24];
  const float* pn_g  = (const float*)d_in[25];
  const float* pn_b  = (const float*)d_in[26];
  const float* ffw1  = (const float*)d_in[27];
  const float* ffb1  = (const float*)d_in[28];
  const float* ffw2  = (const float*)d_in[29];
  const float* ffb2  = (const float*)d_in[30];
  const float* mw1   = (const float*)d_in[31];
  const float* mb1   = (const float*)d_in[32];
  const float* mw2   = (const float*)d_in[33];
  const float* mb2   = (const float*)d_in[34];
  const float* mw3   = (const float*)d_in[35];
  const float* mb3   = (const float*)d_in[36];

  char* ws = (char*)d_ws;
  float*  posT   = (float*)(ws + WS_POST);
  __bf16* wkB    = (__bf16*)(ws + WS_WKB);
  __bf16* wvB    = (__bf16*)(ws + WS_WVB);
  float*  qca    = (float*)(ws + WS_QCA);
  float*  x1     = (float*)(ws + WS_X1);
  __bf16* vhT    = (__bf16*)(ws + WS_VHT);
  float*  logits = (float*)(ws + WS_LOG);
  float*  ca_raw = (float*)(ws + WS_CARAW);
  float*  x4     = (float*)(ws + WS_X4);
  int*    pinds  = (int*)(ws + WS_PIND);
  float*  part   = (float*)(ws + WS_PART);
  float*  out    = (float*)d_out;

  k_pos<<<(Cc * HWp) / 256, 256, 0, stream>>>(posT);
  k_wfrag<<<512, 256, 0, stream>>>(ca_wk, ca_wv, wkB, wvB);
  k_sax1<<<1, 256, 0, stream>>>(sa_wo, sa_bv, sa_bo, ln1_g, ln1_b, x1);
  k_qproj<<<Bz, 256, 0, stream>>>(text, x1, ca_wq, ca_bq, qca);
  k_proj_attn<<<Bz * (HWp / 64), 256, 0, stream>>>(img, posT, wkB, wvB, ca_bk, ca_bv,
                                                   qca, vhT, logits);
  k_softmax<<<Bz * NH * NQ, 256, 0, stream>>>(logits);
  k_avg_gmap<<<(Bz * HWp) / 256, 256, 0, stream>>>(logits, out);
  k_attn_part<<<Bz * SEG, 256, 0, stream>>>(logits, vhT, part);
  k_attn_wo<<<Bz, 256, 0, stream>>>(part, ca_wo, ca_bo, ca_raw);
  k_tail<<<Bz * NQ, 256, 0, stream>>>(ca_raw, x1, ln2_g, ln2_b, ln3_g, ln3_b,
                                      pn_g, pn_b, ffw1, ffb1, ffw2, ffb2, x4);
  k_select<<<Bz, 256, 0, stream>>>(out, pinds);
  k_head<<<Bz * NQ, 256, 0, stream>>>(x4, pinds, img, mw1, mb1, mw2, mb2, mw3, mb3, out);
}
